// Grouped_GEMM_MoE_31799937860088
// MI455X (gfx1250) — matrix operand order repaired and checked by reading, not compiled, not run on hardware
//
#include <hip/hip_runtime.h>
#include <cstdint>
#include <cstddef>

// ---------------- problem constants (match reference) ----------------
#define HIDDEN 2048
#define FFN    2816           // fc1 produces 2*FFN (SwiGLU)
#define NEXP   8
#define TOPK   2
#define NTOK   4096           // BATCH*SEQ
#define TPAIR  (NTOK*TOPK)    // 8192

// ---------------- tiling ----------------
#define BM 64
#define BN 64
#define BK 32                 // K step for v_wmma_f32_16x16x32_bf16
#define LDA 40                // padded LDS row stride in halfs (80B, 16B-multiple)
#define MAXTILES (TPAIR/BM + NEXP)   // 136 worst-case padded row tiles

// ---------------- workspace layout ----------------
#define WS_CNT 0      // [8]
#define WS_CUR 8      // [8]
#define WS_OFF 16     // [9]  exclusive prefix of counts
#define WS_TB  25     // [9]  prefix of ceil(cnt/BM)
#define WS_TOT 34     // [1]
#define WS_PIDS 64    // [TPAIR] pair ids, grouped by expert
#define ACT_OFF ((size_t)64*1024)                          // bf16 [TPAIR][FFN]
#define YP_OFF  (ACT_OFF + (size_t)TPAIR*FFN*2)            // f32  [TPAIR][HIDDEN]

typedef __attribute__((ext_vector_type(16))) __bf16 v16bf;
typedef __attribute__((ext_vector_type(8)))  float  v8f;

struct __align__(16) U4 { unsigned int x, y, z, w; };
union FragB16 { v16bf v; U4 q[2]; };

__device__ __forceinline__ unsigned short f2bf(float f) {
  unsigned int u = __float_as_uint(f);
  u += 0x7FFFu + ((u >> 16) & 1u);      // round-to-nearest-even
  return (unsigned short)(u >> 16);
}

// ---------------- routing ----------------
__global__ void k_init(int* ctrl) {
  int t = threadIdx.x;
  if (t < 16) ctrl[WS_CNT + t] = 0;     // counts + cursors
}

__global__ void k_count(const int* __restrict__ te, int* __restrict__ ctrl) {
  int t = blockIdx.x * blockDim.x + threadIdx.x;
  if (t < TPAIR) atomicAdd(&ctrl[WS_CNT + (te[t] & 7)], 1);
}

__global__ void k_scan(int* ctrl) {
  if (threadIdx.x == 0) {
    int s = 0, tb = 0;
    for (int e = 0; e < NEXP; ++e) {
      ctrl[WS_OFF + e] = s;  s  += ctrl[WS_CNT + e];
      ctrl[WS_TB  + e] = tb; tb += (ctrl[WS_CNT + e] + BM - 1) / BM;
    }
    ctrl[WS_OFF + NEXP] = s;
    ctrl[WS_TB  + NEXP] = tb;
    ctrl[WS_TOT] = tb;
  }
}

__global__ void k_scatter(const int* __restrict__ te, int* __restrict__ ctrl,
                          int* __restrict__ pids) {
  int t = blockIdx.x * blockDim.x + threadIdx.x;
  if (t < TPAIR) {
    int e = te[t] & 7;
    int pos = ctrl[WS_OFF + e] + atomicAdd(&ctrl[WS_CUR + e], 1);
    pids[pos] = t;
  }
}

// ---------------- fc1: x @ w1[e] + b1 -> SwiGLU -> bf16 act ----------------
__global__ __launch_bounds__(128)
void k_fc1(const float* __restrict__ x, const float* __restrict__ w1,
           const float* __restrict__ b1, const int* __restrict__ ctrl,
           const int* __restrict__ pids, unsigned short* __restrict__ act)
{
  const int bx = blockIdx.x;
  if (bx >= ctrl[WS_TOT]) return;
  int e = 0;
  #pragma unroll
  for (int i = 1; i < NEXP; ++i) if (bx >= ctrl[WS_TB + i]) e = i;
  const int row0   = ctrl[WS_OFF + e] + (bx - ctrl[WS_TB + e]) * BM;
  const int rowEnd = ctrl[WS_OFF + e + 1];
  const int n0     = blockIdx.y * BN;

  const int tid  = threadIdx.x;
  const int lane = tid & 31;
  const int wv   = tid >> 5;

  __shared__ __align__(16) unsigned short As[2][BM * LDA];
  __shared__ __align__(16) unsigned short Bs[2][2 * BN * LDA]; // gate rows, then up rows

  const float* w1e = w1 + (size_t)e * HIDDEN * (2 * FFN);
  const float* b1e = b1 + (size_t)e * (2 * FFN);

  v8f acca[4], accb[4];
  #pragma unroll
  for (int j = 0; j < 4; ++j) {
    v8f z = {0.f,0.f,0.f,0.f,0.f,0.f,0.f,0.f};
    acca[j] = z; accb[j] = z;
  }

  const int mrow = wv * 16 + (lane & 15);
  const int kb   = (lane >> 4) << 3;          // unused; A and B both use kb16 (0 or 16)
  const int kb16 = (lane >> 4) << 4;          // B: K base in halfs (0 or 16)

  float4 ar[4], bg[4], bu[4];

  auto loadA = [&](int k0) {
    #pragma unroll
    for (int i = 0; i < 4; ++i) {
      int c = tid + 128 * i;
      int row = c >> 3, kq = (c & 7) << 2;
      float4 v = make_float4(0.f, 0.f, 0.f, 0.f);
      int r = row0 + row;
      if (r < rowEnd) {
        int tok = pids[r] >> 1;               // TOPK == 2
        v = *(const float4*)(x + (size_t)tok * HIDDEN + k0 + kq);
      }
      ar[i] = v;
    }
  };
  auto loadB = [&](int k0) {
    #pragma unroll
    for (int i = 0; i < 4; ++i) {
      int c = tid + 128 * i;
      int k = c >> 4, nq = (c & 15) << 2;
      const float* src = w1e + (size_t)(k0 + k) * (2 * FFN) + n0 + nq;
      bg[i] = *(const float4*)src;
      bu[i] = *(const float4*)(src + FFN);
    }
  };
  auto storeTile = [&](int buf) {
    unsigned short* Ab = &As[buf][0];
    unsigned short* Bb = &Bs[buf][0];
    #pragma unroll
    for (int i = 0; i < 4; ++i) {
      int c = tid + 128 * i;
      int row = c >> 3, kq = (c & 7) << 2;
      unsigned short* d = &Ab[row * LDA + kq];
      d[0] = f2bf(ar[i].x); d[1] = f2bf(ar[i].y);
      d[2] = f2bf(ar[i].z); d[3] = f2bf(ar[i].w);
    }
    #pragma unroll
    for (int i = 0; i < 4; ++i) {
      int c = tid + 128 * i;
      int k = c >> 4, nq = (c & 15) << 2;
      Bb[(nq + 0) * LDA + k] = f2bf(bg[i].x);
      Bb[(nq + 1) * LDA + k] = f2bf(bg[i].y);
      Bb[(nq + 2) * LDA + k] = f2bf(bg[i].z);
      Bb[(nq + 3) * LDA + k] = f2bf(bg[i].w);
      Bb[(BN + nq + 0) * LDA + k] = f2bf(bu[i].x);
      Bb[(BN + nq + 1) * LDA + k] = f2bf(bu[i].y);
      Bb[(BN + nq + 2) * LDA + k] = f2bf(bu[i].z);
      Bb[(BN + nq + 3) * LDA + k] = f2bf(bu[i].w);
    }
  };
  auto compute = [&](int buf) {
    const unsigned short* Ab = &As[buf][0];
    const unsigned short* Bb = &Bs[buf][0];
    FragB16 af;
    af.q[0] = *(const U4*)&Ab[mrow * LDA + kb16];
    af.q[1] = *(const U4*)&Ab[mrow * LDA + kb16 + 8];
    FragB16 bfv[8];
    #pragma unroll
    for (int j = 0; j < 4; ++j) {
      int nl = j * 16 + (lane & 15);
      bfv[2*j  ].q[0] = *(const U4*)&Bb[nl * LDA + kb16];
      bfv[2*j  ].q[1] = *(const U4*)&Bb[nl * LDA + kb16 + 8];
      bfv[2*j+1].q[0] = *(const U4*)&Bb[(BN + nl) * LDA + kb16];
      bfv[2*j+1].q[1] = *(const U4*)&Bb[(BN + nl) * LDA + kb16 + 8];
    }
    #pragma unroll
    for (int j = 0; j < 4; ++j) {
      acca[j] = __builtin_amdgcn_wmma_f32_16x16x32_bf16(false, af.v, false, bfv[2*j].v,
                                                        (short)0, acca[j], false, false);
      accb[j] = __builtin_amdgcn_wmma_f32_16x16x32_bf16(false, af.v, false, bfv[2*j+1].v,
                                                        (short)0, accb[j], false, false);
    }
  };

  // software pipeline: global loads for tile k+1 in flight while tile k runs WMMA
  loadA(0); loadB(0);
  storeTile(0);
  __syncthreads();
  int cur = 0;
  for (int k0 = 0; k0 < HIDDEN; k0 += BK) {
    const bool has_next = (k0 + BK) < HIDDEN;
    if (has_next) {
      __builtin_prefetch(w1e + (size_t)(k0 + BK) * (2 * FFN) + n0 + ((tid & 15) << 2), 0, 1);
      loadA(k0 + BK); loadB(k0 + BK);
    }
    compute(cur);
    if (has_next) storeTile(cur ^ 1);
    __syncthreads();
    cur ^= 1;
  }

  // ---- epilogue: +bias, SwiGLU, store bf16 activation at permuted row ----
  #pragma unroll
  for (int j = 0; j < 4; ++j) {
    int cc = n0 + j * 16 + (lane & 15);
    float bga = b1e[cc], bub = b1e[FFN + cc];
    #pragma unroll
    for (int jj = 0; jj < 8; ++jj) {
      int r = row0 + wv * 16 + jj + ((lane >> 4) << 3);
      if (r < rowEnd) {
        float a = acca[j][jj] + bga;
        float b = accb[j][jj] + bub;
        float s = a / (1.f + __expf(-a));   // silu(a)
        act[(size_t)r * FFN + cc] = f2bf(s * b);
      }
    }
  }
}

// ---------------- fc2: act @ w2[e] + b2, scaled by expert weight ----------------
__global__ __launch_bounds__(128)
void k_fc2(const unsigned short* __restrict__ act, const float* __restrict__ w2,
           const float* __restrict__ b2, const float* __restrict__ ew,
           const int* __restrict__ ctrl, const int* __restrict__ pids,
           float* __restrict__ yp)
{
  const int bx = blockIdx.x;
  if (bx >= ctrl[WS_TOT]) return;
  int e = 0;
  #pragma unroll
  for (int i = 1; i < NEXP; ++i) if (bx >= ctrl[WS_TB + i]) e = i;
  const int row0   = ctrl[WS_OFF + e] + (bx - ctrl[WS_TB + e]) * BM;
  const int rowEnd = ctrl[WS_OFF + e + 1];
  const int n0     = blockIdx.y * BN;

  const int tid  = threadIdx.x;
  const int lane = tid & 31;
  const int wv   = tid >> 5;

  __shared__ __align__(16) unsigned short As[2][BM * LDA];
  __shared__ __align__(16) unsigned short Bs[2][BN * LDA];

  const float* w2e = w2 + (size_t)e * FFN * HIDDEN;
  const float* b2e = b2 + (size_t)e * HIDDEN;

  v8f acc[4];
  #pragma unroll
  for (int j = 0; j < 4; ++j) {
    v8f z = {0.f,0.f,0.f,0.f,0.f,0.f,0.f,0.f};
    acc[j] = z;
  }

  const int mrow = wv * 16 + (lane & 15);
  const int kb   = (lane >> 4) << 3;
  const int kb16 = (lane >> 4) << 4;

  float4 br[4];

  // A tile: bf16 activations go straight to LDS via the gfx1250 async path
  auto asyncLoadA = [&](int k0, int buf) {
    #pragma unroll
    for (int i = 0; i < 2; ++i) {
      int c = tid + 128 * i;
      int row = c >> 2, kq8 = (c & 3) << 3;
      int r = row0 + row;
      if (r < rowEnd) {
        const unsigned short* g = act + (size_t)r * FFN + k0 + kq8;
        unsigned lds = (unsigned)(size_t)&As[buf][row * LDA + kq8];
        asm volatile("global_load_async_to_lds_b128 %0, %1, off"
                     :: "v"(lds), "v"(g) : "memory");
      }
    }
  };
  auto loadB = [&](int k0) {
    #pragma unroll
    for (int i = 0; i < 4; ++i) {
      int c = tid + 128 * i;
      int k = c >> 4, nq = (c & 15) << 2;
      br[i] = *(const float4*)(w2e + (size_t)(k0 + k) * HIDDEN + n0 + nq);
    }
  };
  auto storeB = [&](int buf) {
    unsigned short* Bb = &Bs[buf][0];
    #pragma unroll
    for (int i = 0; i < 4; ++i) {
      int c = tid + 128 * i;
      int k = c >> 4, nq = (c & 15) << 2;
      Bb[(nq + 0) * LDA + k] = f2bf(br[i].x);
      Bb[(nq + 1) * LDA + k] = f2bf(br[i].y);
      Bb[(nq + 2) * LDA + k] = f2bf(br[i].z);
      Bb[(nq + 3) * LDA + k] = f2bf(br[i].w);
    }
  };
  auto compute = [&](int buf) {
    const unsigned short* Ab = &As[buf][0];
    const unsigned short* Bb = &Bs[buf][0];
    FragB16 af;
    af.q[0] = *(const U4*)&Ab[mrow * LDA + kb16];
    af.q[1] = *(const U4*)&Ab[mrow * LDA + kb16 + 8];
    FragB16 bfv[4];
    #pragma unroll
    for (int j = 0; j < 4; ++j) {
      int nl = j * 16 + (lane & 15);
      bfv[j].q[0] = *(const U4*)&Bb[nl * LDA + kb16];
      bfv[j].q[1] = *(const U4*)&Bb[nl * LDA + kb16 + 8];
    }
    #pragma unroll
    for (int j = 0; j < 4; ++j)
      acc[j] = __builtin_amdgcn_wmma_f32_16x16x32_bf16(false, af.v, false, bfv[j].v,
                                                       (short)0, acc[j], false, false);
  };

  asyncLoadA(0, 0);
  loadB(0);
  storeB(0);
  asm volatile("s_wait_asynccnt 0x0" ::: "memory");
  __syncthreads();
  int cur = 0;
  for (int k0 = 0; k0 < FFN; k0 += BK) {
    const bool has_next = (k0 + BK) < FFN;
    if (has_next) {
      __builtin_prefetch(w2e + (size_t)(k0 + BK) * HIDDEN + n0 + ((tid & 15) << 2), 0, 1);
      asyncLoadA(k0 + BK, cur ^ 1);
      loadB(k0 + BK);
    }
    compute(cur);
    if (has_next) storeB(cur ^ 1);
    asm volatile("s_wait_asynccnt 0x0" ::: "memory");
    __syncthreads();
    cur ^= 1;
  }

  // ---- epilogue: +b2, scale by routing weight, scatter to pair-order rows ----
  #pragma unroll
  for (int j = 0; j < 4; ++j) {
    int cc = n0 + j * 16 + (lane & 15);
    float bias = b2e[cc];
    #pragma unroll
    for (int jj = 0; jj < 8; ++jj) {
      int r = row0 + wv * 16 + jj + ((lane >> 4) << 3);
      if (r < rowEnd) {
        int t = pids[r];
        yp[(size_t)t * HIDDEN + cc] = ew[t] * (acc[j][jj] + bias);
      }
    }
  }
}

// ---------------- combine the two topk slots per token ----------------
__global__ void k_combine(const float* __restrict__ yp, float* __restrict__ out) {
  const int QH = HIDDEN / 4;
  int g = blockIdx.x * blockDim.x + threadIdx.x;
  if (g < NTOK * QH) {
    int n = g / QH, q = g - n * QH;
    const float4* p = (const float4*)yp;
    float4 a = p[(size_t)(2 * n) * QH + q];
    float4 b = p[(size_t)(2 * n + 1) * QH + q];
    float4 o;
    o.x = a.x + b.x; o.y = a.y + b.y; o.z = a.z + b.z; o.w = a.w + b.w;
    ((float4*)out)[g] = o;
  }
}

// ---------------- host entry ----------------
extern "C" void kernel_launch(void* const* d_in, const int* in_sizes, int n_in,
                              void* d_out, int out_size, void* d_ws, size_t ws_size,
                              hipStream_t stream) {
  (void)in_sizes; (void)n_in; (void)out_size; (void)ws_size;
  const float* x  = (const float*)d_in[0];   // hidden_states
  const float* ew = (const float*)d_in[1];   // expert_weights
  const float* w1 = (const float*)d_in[2];
  const float* b1 = (const float*)d_in[3];
  const float* w2 = (const float*)d_in[4];
  const float* b2 = (const float*)d_in[5];
  const int*   te = (const int*)d_in[6];     // top_experts
  float* out = (float*)d_out;

  char* ws = (char*)d_ws;
  int*  ctrl = (int*)ws;
  int*  pids = ctrl + WS_PIDS;
  unsigned short* act = (unsigned short*)(ws + ACT_OFF);
  float* yp = (float*)(ws + YP_OFF);

  k_init   <<<1, 32, 0, stream>>>(ctrl);
  k_count  <<<TPAIR / 256, 256, 0, stream>>>(te, ctrl);
  k_scan   <<<1, 1, 0, stream>>>(ctrl);
  k_scatter<<<TPAIR / 256, 256, 0, stream>>>(te, ctrl, pids);
  k_fc1    <<<dim3(MAXTILES, FFN / BN),    128, 0, stream>>>(x, w1, b1, ctrl, pids, act);
  k_fc2    <<<dim3(MAXTILES, HIDDEN / BN), 128, 0, stream>>>(act, w2, b2, ew, ctrl, pids, yp);
  k_combine<<<(NTOK * (HIDDEN / 4)) / 256, 256, 0, stream>>>(yp, out);
}
